// AlphaQubitDecoder_73100343378496
// MI455X (gfx1250) — matrix operand order repaired and checked by reading, not compiled, not run on hardware
//
#include <hip/hip_runtime.h>
#include <math.h>

// ---------------------------------------------------------------- types
typedef _Float16 h16;
typedef __attribute__((ext_vector_type(16))) _Float16 v16h;
typedef __attribute__((ext_vector_type(8)))  float    v8f;

// ---------------------------------------------------------------- model constants
#define Bsz   64
#define Rr    10
#define Gg    11
#define Ss    121          // G*G
#define NFf   5
#define HIDc  256
#define HEADSc 8
#define HDc   32
#define PEc   48
#define MROWS (Bsz*Ss)     // 7744 = 484*16

// =================================================================
// Generic WMMA GEMM:  C[M,N] = gather(A)[M,K] * Bpk[K,N] + bias, epilogue
// MODE 0: A is [M,K] row-major f32
// MODE 1: implicit dilated 3x3 conv gather, A is state [B*S, 256], K = 27*256
// MODE 2: implicit readout 2x2 conv gather (front zero-pad 23), K = 4*256
// EPI  0: none   1: relu   2: glu  (C = (acc+bias) * sigmoid(aux))
// block = 128 threads (4 waves); block tile = 16 rows x 256 cols
// Double-buffered A tile in LDS, one barrier per k-iteration;
// B weights prepacked into the exact wave32 B-fragment layout (v16h loads).
// =================================================================
template<int MODE, int EPI>
__global__ __launch_bounds__(128)
void gemm_k(const float* __restrict__ A, const h16* __restrict__ Bpk,
            const float* __restrict__ bias, const float* __restrict__ aux,
            float* __restrict__ C, int M, int N, int K)
{
    __shared__ __align__(32) h16 At[2][16*32];
    const int tid  = threadIdx.x;
    const int lane = tid & 31;
    const int wave = tid >> 5;
    const int mBase   = blockIdx.x * 16;
    const int colBase = blockIdx.y * 256 + wave * 64;
    const int m  = lane & 15;
    const int hi = lane >> 4;
    const int kTiles = K >> 5;
    const int nTiles = N >> 4;

    // ---- loop-invariant gather state for this thread's A-tile slot
    const int r  = (tid * 4) >> 5;     // row within 16-row tile
    const int c0 = (tid * 4) & 31;     // first of 4 consecutive k-cols
    const int grow = mBase + r;
    int gb = 0, gy = 0, gx = 0;
    if (MODE != 0) {
        gb = grow / Ss;
        int gs = grow - gb * Ss;
        gy = gs / Gg;
        gx = gs - gy * Gg;
    }
    const float* Arow = (MODE == 0) ? (A + (size_t)grow * K) : A;

    // gather 4 consecutive A elements for k-tile kt (tap is tile-invariant: 32 | 256)
    auto gather4 = [&](int kt, float v[4]) {
        if (MODE == 0) {
            const float* p = Arow + kt*32 + c0;
            v[0] = p[0]; v[1] = p[1]; v[2] = p[2]; v[3] = p[3];
        } else if (MODE == 1) {
            int tap = kt >> 3;                       // 8 k-tiles per tap
            int kin = ((kt & 7) << 5) + c0;
            int t0  = tap / 9, kk = tap - t0*9;
            int d   = 1 << t0;                       // dilation 1,2,4
            int rr  = gy + d * (kk/3 - 1);
            int cc  = gx + d * (kk%3 - 1);
            if (rr >= 0 && rr < Gg && cc >= 0 && cc < Gg) {
                const float* p = Arow + ((size_t)(gb*Ss + rr*Gg + cc))*HIDc + kin;
                v[0] = p[0]; v[1] = p[1]; v[2] = p[2]; v[3] = p[3];
            } else {
                v[0] = v[1] = v[2] = v[3] = 0.f;
            }
        } else {                                      // MODE 2
            int tap = kt >> 3;
            int kin = ((kt & 7) << 5) + c0;
            int ky  = tap >> 1, kx = tap & 1;
            int pp  = (gy + ky)*(Gg+1) + (gx + kx);   // 12x12 padded grid
            int q   = pp - 23;                        // front zero-pad
            if (q >= 0) {
                const float* p = Arow + ((size_t)(gb*Ss + q))*HIDc + kin;
                v[0] = p[0]; v[1] = p[1]; v[2] = p[2]; v[3] = p[3];
            } else {
                v[0] = v[1] = v[2] = v[3] = 0.f;
            }
        }
    };

    v8f acc[4];
    acc[0] = v8f{}; acc[1] = v8f{}; acc[2] = v8f{}; acc[3] = v8f{};

    // ---- preload k-tile 0
    {
        float v[4];
        gather4(0, v);
        #pragma unroll
        for (int j = 0; j < 4; ++j) At[0][r*32 + c0 + j] = (h16)v[j];
    }
    __syncthreads();

    for (int kt = 0; kt < kTiles; ++kt) {
        const int cur = kt & 1;
        // issue next tile's global loads early (hide behind WMMA)
        float nv[4];
        const bool hasNext = (kt + 1 < kTiles);
        if (hasNext) gather4(kt + 1, nv);
        // prefetch B-fragment stream two tiles ahead (global_prefetch_b8)
        if (kt + 2 < kTiles)
            __builtin_prefetch((const void*)(Bpk +
                (((size_t)(kt + 2) * nTiles + (colBase >> 4)) * 32 + lane) * 16), 0, 1);

        // ---- A fragment (16x32 f16): lane m, VGPR halves -> k
        union { v16h v; unsigned u[8]; } af;
        #pragma unroll
        for (int p = 0; p < 8; ++p) {
            int k0 = p*2 + hi*16;
            af.u[p] = *(const unsigned*)&At[cur][m*32 + k0];
        }
        // ---- 4 prepacked B fragments + WMMA
        const v16h* bp = (const v16h*)(Bpk +
            (((size_t)kt * nTiles + (colBase >> 4)) * 32 + lane) * 16);
        #pragma unroll
        for (int nf = 0; nf < 4; ++nf) {
            v16h bf = bp[nf * 32];   // next n-tile = +512 halves = +32 v16h
            acc[nf] = __builtin_amdgcn_wmma_f32_16x16x32_f16(
                false, af.v, false, bf, (short)0, acc[nf], false, false);
        }
        // ---- commit next tile to the other LDS buffer
        if (hasNext) {
            #pragma unroll
            for (int j = 0; j < 4; ++j) At[cur ^ 1][r*32 + c0 + j] = (h16)nv[j];
        }
        __syncthreads();
    }

    // ---- epilogue
    #pragma unroll
    for (int nf = 0; nf < 4; ++nf) {
        int col = colBase + nf*16 + m;
        float bv = bias ? bias[col] : 0.f;
        #pragma unroll
        for (int i = 0; i < 8; ++i) {
            int row = mBase + i + hi*8;
            float v = acc[nf][i] + bv;
            if (EPI == 1) v = v > 0.f ? v : 0.f;
            if (EPI == 2) {
                float gv = aux[(size_t)row*N + col];
                v = v * (1.f / (1.f + __expf(-gv)));
            }
            C[(size_t)row*N + col] = v;
        }
    }
}

// =================================================================
// Weight repacking into B-fragment layout:
//  dst[((k/32)*(N/16)+(n/16))*32 + lane][half], lane=(n&15)+16*((k>>4)&1), half=k&15
// =================================================================
__device__ __forceinline__ size_t pack_idx(int k, int n, int N) {
    return (((size_t)(k >> 5)*(N >> 4) + (n >> 4))*32
            + ((n & 15) + ((k >> 4) & 1)*16))*16 + (k & 15);
}

__global__ void repack_kn_k(const float* __restrict__ src, h16* __restrict__ dst,
                            int K, int N)
{
    int t = blockIdx.x*256 + threadIdx.x;
    if (t >= K*N) return;
    int k = t / N, n = t - k*N;
    dst[pack_idx(k, n, N)] = (h16)src[t];
}

__global__ void repack_conv_k(const float* __restrict__ w, h16* __restrict__ dst)
{
    // conv_w (3,256,256,3,3) [dil][out][in][ky][kx] -> K=6912, N=256
    const int N = 256;
    int t = blockIdx.x*256 + threadIdx.x;
    if (t >= 6912*256) return;
    int k = t / N, n = t - k*N;
    int tap = k >> 8, kin = k & 255;
    int t0 = tap / 9, kk = tap - t0*9;
    int ky = kk / 3, kx = kk - ky*3;
    float v = w[((((size_t)t0*HIDc + n)*HIDc + kin)*3 + ky)*3 + kx];
    dst[pack_idx(k, n, N)] = (h16)v;
}

__global__ void repack_roconv_k(const float* __restrict__ w, h16* __restrict__ dst)
{
    // readout conv_w (256,256,2,2) [out][in][ky][kx] -> K=1024, N=256
    const int N = 256;
    int t = blockIdx.x*256 + threadIdx.x;
    if (t >= 1024*256) return;
    int k = t / N, n = t - k*N;
    int tap = k >> 8, kin = k & 255;
    int ky = tap >> 1, kx = tap & 1;
    float v = w[(((size_t)n*HIDc + kin)*2 + ky)*2 + kx];
    dst[pack_idx(k, n, N)] = (h16)v;
}

// =================================================================
// Attention: one block (128 thr, 4 waves) per (batch, head)
// =================================================================
__global__ __launch_bounds__(128)
void attn_k(const float* __restrict__ qkv, const float* __restrict__ evt,
            const float* __restrict__ pev, const float* __restrict__ sbh,
            const float* __restrict__ Wb,  const float* __restrict__ bb,
            float* __restrict__ out)
{
    __shared__ __align__(32) h16 qs[128*32];
    __shared__ __align__(32) h16 ks[128*32];
    __shared__ __align__(32) h16 vT[32*128];
    __shared__ __align__(32) h16 sc[128*128];
    __shared__ float ev[128];
    __shared__ float pv[128];

    const int tid  = threadIdx.x;
    const int lane = tid & 31;
    const int wave = tid >> 5;
    const int bh = blockIdx.x;
    const int b  = bh >> 3;
    const int h  = bh & 7;

    // dynamic-bias weights (rows 48..54 of Wb [55,8]) + bias bb[h]
    float w0 = Wb[48*8 + h], w1 = Wb[49*8 + h], w2 = Wb[50*8 + h];
    float w3 = Wb[51*8 + h], w4 = Wb[52*8 + h], w5 = Wb[53*8 + h];
    float w6 = Wb[54*8 + h], bbh = bb[h];

    // load q, k, v (pad tokens 121..127 with zero); v stored transposed
    for (int t = tid; t < 128*32; t += 128) {
        int tok = t >> 5, d = t & 31;
        float qv = 0.f, kv = 0.f, vv = 0.f;
        if (tok < Ss) {
            size_t base = ((size_t)(b*Ss + tok))*768;
            qv = qkv[base +        h*HDc + d];
            kv = qkv[base + 256  + h*HDc + d];
            vv = qkv[base + 512  + h*HDc + d];
        }
        qs[t] = (h16)qv; ks[t] = (h16)kv; vT[d*128 + tok] = (h16)vv;
    }
    {
        int i = tid;
        ev[i] = (i < Ss) ? evt[b*Ss + i] : 0.f;
        pv[i] = (i < Ss) ? pev[b*Ss + i] : 0.f;
    }
    __syncthreads();

    const int m  = lane & 15;
    const int hi = lane >> 4;
    const float scale = 0.1767766952966369f;   // 1/sqrt(32)

    // ---- scores = q k^T * scale + bias  -> f16 in LDS
    for (int itl = 0; itl < 2; ++itl) {
        int it   = wave*2 + itl;
        int rowm = it*16 + m;
        union { v16h v; unsigned u[8]; } af;
        #pragma unroll
        for (int p = 0; p < 8; ++p) {
            int k0 = p*2 + hi*16;
            af.u[p] = *(const unsigned*)&qs[rowm*32 + k0];
        }
        for (int jt = 0; jt < 8; ++jt) {
            int j = jt*16 + m;
            v16h bf = *(const v16h*)&ks[j*32 + hi*16];
            v8f c = v8f{};
            c = __builtin_amdgcn_wmma_f32_16x16x32_f16(
                    false, af.v, false, bf, (short)0, c, false, false);
            float evj = ev[j], pvj = pv[j];
            #pragma unroll
            for (int i = 0; i < 8; ++i) {
                int row = it*16 + i + hi*8;
                float val;
                if (j < Ss) {
                    float evi = ev[row], pvi = pv[row];
                    val = c[i]*scale + bbh
                        + evi*evj*w0 + evi*pvj*w1 + pvi*evj*w2 + pvi*pvj*w3;
                    if (row == j) val += evi*w4 + pvi*w5 + evi*pvi*w6;
                    if (row < Ss) val += sbh[((size_t)(row*Ss + j))*8 + h];
                } else {
                    val = -30000.f;   // mask padded keys
                }
                sc[row*128 + j] = (h16)val;
            }
        }
    }
    __syncthreads();

    // ---- row softmax (thread r handles row r)
    {
        int r = tid;
        float mx = -1e30f;
        for (int j = 0; j < Ss; ++j) {
            float v = (float)sc[r*128 + j];
            mx = v > mx ? v : mx;
        }
        float sum = 0.f;
        for (int j = 0; j < Ss; ++j) sum += __expf((float)sc[r*128 + j] - mx);
        float inv = 1.f / sum;
        for (int j = 0; j < 128; ++j) {
            float v = (j < Ss) ? __expf((float)sc[r*128 + j] - mx) * inv : 0.f;
            sc[r*128 + j] = (h16)v;
        }
    }
    __syncthreads();

    // ---- out = attn @ v
    for (int itl = 0; itl < 2; ++itl) {
        int it   = wave*2 + itl;
        int rowm = it*16 + m;
        for (int nt = 0; nt < 2; ++nt) {
            v8f c = v8f{};
            #pragma unroll
            for (int kt = 0; kt < 4; ++kt) {
                union { v16h v; unsigned u[8]; } af;
                #pragma unroll
                for (int p = 0; p < 8; ++p) {
                    int k0 = kt*32 + p*2 + hi*16;
                    af.u[p] = *(const unsigned*)&sc[rowm*128 + k0];
                }
                v16h bf = *(const v16h*)&vT[(nt*16 + m)*128 + kt*32 + hi*16];
                c = __builtin_amdgcn_wmma_f32_16x16x32_f16(
                        false, af.v, false, bf, (short)0, c, false, false);
            }
            #pragma unroll
            for (int i = 0; i < 8; ++i) {
                int row = it*16 + i + hi*8;
                if (row < Ss)
                    out[((size_t)(b*Ss + row))*HIDc + h*HDc + nt*16 + m] = c[i];
            }
        }
    }
}

// =================================================================
// LayerNorm (optionally with residual):  out = LN(x (+res)) * g + b
// =================================================================
__global__ __launch_bounds__(256)
void ln_k(const float* __restrict__ x, const float* __restrict__ res,
          const float* __restrict__ g, const float* __restrict__ bta,
          float* __restrict__ out)
{
    __shared__ float sd[256];
    int row = blockIdx.x, d = threadIdx.x;
    float v = x[(size_t)row*HIDc + d];
    if (res) v += res[(size_t)row*HIDc + d];
    sd[d] = v; __syncthreads();
    for (int s = 128; s > 0; s >>= 1) { if (d < s) sd[d] += sd[d+s]; __syncthreads(); }
    float mean = sd[0] * (1.f/256.f); __syncthreads();
    float c = v - mean;
    sd[d] = c*c; __syncthreads();
    for (int s = 128; s > 0; s >>= 1) { if (d < s) sd[d] += sd[d+s]; __syncthreads(); }
    float var = sd[0] * (1.f/256.f);
    out[(size_t)row*HIDc + d] = c * rsqrtf(var + 1e-5f) * g[d] + bta[d];
}

// =================================================================
// Embedding + LN + recurrent state update; also extracts evt
// =================================================================
__global__ __launch_bounds__(256)
void embed_k(const float* __restrict__ inp, const int* __restrict__ fmask,
             const float* __restrict__ Wf, const float* __restrict__ bf,
             const float* __restrict__ idxe, const float* __restrict__ onv,
             const float* __restrict__ offv, const float* __restrict__ g,
             const float* __restrict__ bta, float* __restrict__ state,
             float* __restrict__ evtCur, int r)
{
    __shared__ float sd[256];
    __shared__ float xs[NFf];
    int row = blockIdx.x, d = threadIdx.x;
    int b = row / Ss, s = row - b*Ss;
    size_t xb = ((size_t)(b*Rr + r)*Ss + s)*NFf;
    if (d < NFf) xs[d] = inp[xb + d];
    __syncthreads();
    int fm = (r == Rr-1) ? fmask[row] : 0;
    float hsum = idxe[s*HIDc + d];
    #pragma unroll
    for (int f = 0; f < NFf; ++f) hsum += xs[f]*Wf[f*HIDc + d] + bf[f*HIDc + d];
    if (fm == 1) hsum += onv[d];
    if (fm == 2) hsum += offv[d];
    sd[d] = hsum; __syncthreads();
    for (int t = 128; t > 0; t >>= 1) { if (d < t) sd[d] += sd[d+t]; __syncthreads(); }
    float mean = sd[0] * (1.f/256.f); __syncthreads();
    float c = hsum - mean; sd[d] = c*c; __syncthreads();
    for (int t = 128; t > 0; t >>= 1) { if (d < t) sd[d] += sd[d+t]; __syncthreads(); }
    float var = sd[0] * (1.f/256.f);
    float y = c * rsqrtf(var + 1e-5f) * g[d] + bta[d];
    state[(size_t)row*HIDc + d] =
        (state[(size_t)row*HIDc + d] + y) * 0.7071067811865475f;
    if (d == 0) evtCur[row] = xs[0];
}

// =================================================================
// Static pairwise bias table: sbh[i,j,h]  (per layer)
// =================================================================
__global__ __launch_bounds__(64)
void sbh_k(const float* __restrict__ rowE, const float* __restrict__ colE,
           const float* __restrict__ dxE,  const float* __restrict__ dyE,
           const float* __restrict__ manE, const float* __restrict__ sameE,
           const float* __restrict__ mW1,  const float* __restrict__ mb1,
           const float* __restrict__ mW2,  const float* __restrict__ mb2,
           const float* __restrict__ Wb,   float* __restrict__ sbh)
{
    __shared__ float e[PEc], h1[PEc], h2[PEc];
    int ij = blockIdx.x;
    int i = ij / Ss, j = ij - i*Ss;
    int t = threadIdx.x;
    int ri = i / Gg, ci = i - ri*Gg, rj = j / Gg, cj = j - rj*Gg;
    if (t < PEc) {
        int grp = t / 6, cm = t % 6;
        int dr = ri > rj ? ri - rj : rj - ri;
        int dc = ci > cj ? ci - cj : cj - ci;
        float v;
        switch (grp) {
            case 0: v = rowE[ri*6 + cm]; break;
            case 1: v = colE[ci*6 + cm]; break;
            case 2: v = rowE[rj*6 + cm]; break;
            case 3: v = colE[cj*6 + cm]; break;
            case 4: v = dxE[(ri - rj + Gg)*6 + cm]; break;
            case 5: v = dyE[(ci - cj + Gg)*6 + cm]; break;
            case 6: v = manE[(dr + dc)*6 + cm]; break;
            default: v = sameE[((((ri+ci)&1) == ((rj+cj)&1)) ? 1 : 0)*6 + cm]; break;
        }
        e[t] = v;
    }
    __syncthreads();
    if (t < PEc) {
        float a = mb1[t];
        for (int p = 0; p < PEc; ++p) a += e[p]*mW1[p*PEc + t];
        h1[t] = a > 0.f ? a : 0.f;
    }
    __syncthreads();
    if (t < PEc) {
        float a = mb2[t];
        for (int p = 0; p < PEc; ++p) a += h1[p]*mW2[p*PEc + t];
        h2[t] = a;
    }
    __syncthreads();
    if (t < HEADSc) {
        float a = 0.f;
        for (int p = 0; p < PEc; ++p) a += h2[p]*Wb[p*HEADSc + t];
        sbh[(size_t)ij*HEADSc + t] = a;
    }
}

// ---------------------------------------------------------------- small utility kernels
__global__ void zero_k(float* p, int n) {
    int t = blockIdx.x*256 + threadIdx.x;
    if (t < n) p[t] = 0.f;
}

__global__ void convbias_k(const float* __restrict__ cb, float* __restrict__ out) {
    int d = threadIdx.x;
    out[d] = cb[d] + cb[256 + d] + cb[512 + d];
}

__global__ __launch_bounds__(256)
void lines_k(const float* __restrict__ df, const int* __restrict__ basis,
             float* __restrict__ lines)
{
    int blk = blockIdx.x;
    int b = blk / Gg, l = blk - b*Gg;
    int d = threadIdx.x;
    int bs = basis[b];
    float s = 0.f;
    for (int x = 0; x < Gg; ++x) {
        int tok = (bs == 0) ? (l*Gg + x) : (x*Gg + l);
        s += df[((size_t)(b*Ss + tok))*HIDc + d];
    }
    lines[(size_t)blk*HIDc + d] = s * (1.f/11.f);
}

__global__ __launch_bounds__(256)
void final_k(const float* __restrict__ y1, const float* __restrict__ W2,
             const float* __restrict__ b2, float* __restrict__ out)
{
    __shared__ float sd[256];
    int b = blockIdx.x, d = threadIdx.x;
    float s = 0.f;
    for (int l = 0; l < Gg; ++l) s += y1[((size_t)(b*Gg + l))*HIDc + d];
    sd[d] = s * W2[d]; __syncthreads();
    for (int t = 128; t > 0; t >>= 1) { if (d < t) sd[d] += sd[d+t]; __syncthreads(); }
    if (d == 0) out[b] = sd[0]*(1.f/11.f) + b2[0];
}

// =================================================================
// Host orchestration
// =================================================================
extern "C" void kernel_launch(void* const* d_in, const int* in_sizes, int n_in,
                              void* d_out, int out_size, void* d_ws, size_t ws_size,
                              hipStream_t stream)
{
    (void)in_sizes; (void)n_in; (void)out_size; (void)ws_size;
    auto P = [&](int i) { return (const float*)d_in[i]; };
    auto PI = [&](int i) { return (const int*)d_in[i]; };

    // ---- input layout (setup_inputs dict order, recursively) ----
    const float* inputs = P(0);
    const int*   basis  = PI(1);
    const int*   fmask  = PI(2);
    // emb: 3..9 = Wf, bf, idx, on, off, ln_g, ln_b
    // layer l base = 10 + 30*l (see comment block in round 1)
    const int RO = 10 + 30*3;   // readout base: conv_w, conv_b, W1, b1, W2, b2

    // ---- workspace carve-up ----
    char* ws = (char*)d_ws;
    size_t off = 0;
    auto alloc = [&](size_t bytes) -> void* {
        void* p = ws + off;
        off = (off + bytes + 255) & ~(size_t)255;
        return p;
    };
    float* state = (float*)alloc((size_t)MROWS*HIDc*4);
    float* bufS  = (float*)alloc((size_t)MROWS*HIDc*4);
    float* bufS2 = (float*)alloc((size_t)MROWS*HIDc*4);
    float* bufT  = (float*)alloc((size_t)MROWS*HIDc*4);
    float* bufU  = (float*)alloc((size_t)MROWS*HIDc*4);
    float* qkvb  = (float*)alloc((size_t)MROWS*768*4);
    float* gbuf  = (float*)alloc((size_t)MROWS*1024*4);
    float* hbuf  = (float*)alloc((size_t)MROWS*1024*4);
    float* evtA  = (float*)alloc((size_t)MROWS*4);
    float* evtB  = (float*)alloc((size_t)MROWS*4);
    float* sbh   = (float*)alloc((size_t)3*Ss*Ss*HEADSc*4);
    float* cbias = (float*)alloc((size_t)3*HIDc*4);
    h16 *pk_conv[3], *pk_qkv[3], *pk_wo[3], *pk_wp[3], *pk_wg[3], *pk_wout[3];
    for (int l = 0; l < 3; ++l) {
        pk_conv[l] = (h16*)alloc((size_t)6912*256*2);
        pk_qkv[l]  = (h16*)alloc((size_t)256*768*2);
        pk_wo[l]   = (h16*)alloc((size_t)256*256*2);
        pk_wp[l]   = (h16*)alloc((size_t)256*1024*2);
        pk_wg[l]   = (h16*)alloc((size_t)256*1024*2);
        pk_wout[l] = (h16*)alloc((size_t)1024*256*2);
    }
    h16* pk_roconv = (h16*)alloc((size_t)1024*256*2);
    h16* pk_w1     = (h16*)alloc((size_t)256*256*2);
    float* linesb  = (float*)alloc((size_t)Bsz*Gg*HIDc*4);
    float* y1b     = (float*)alloc((size_t)Bsz*Gg*HIDc*4);

    // ---- init ----
    zero_k<<<(MROWS*HIDc + 255)/256, 256, 0, stream>>>(state, MROWS*HIDc);
    zero_k<<<(MROWS + 255)/256, 256, 0, stream>>>(evtB, MROWS);

    // ---- per-layer precompute: conv bias sum, static bias table, weight repack ----
    for (int l = 0; l < 3; ++l) {
        int LB = 10 + 30*l;
        convbias_k<<<1, 256, 0, stream>>>(P(LB+1), cbias + l*HIDc);
        sbh_k<<<Ss*Ss, 64, 0, stream>>>(P(LB+2), P(LB+3), P(LB+4), P(LB+5),
                                        P(LB+6), P(LB+7), P(LB+8), P(LB+9),
                                        P(LB+10), P(LB+11), P(LB+16),
                                        sbh + (size_t)l*Ss*Ss*HEADSc);
        repack_conv_k<<<6912, 256, 0, stream>>>(P(LB+0), pk_conv[l]);
        repack_kn_k<<<(256*768)/256, 256, 0, stream>>>(P(LB+12), pk_qkv[l], 256, 768);
        repack_kn_k<<<(256*256)/256, 256, 0, stream>>>(P(LB+14), pk_wo[l],  256, 256);
        repack_kn_k<<<(256*1024)/256, 256, 0, stream>>>(P(LB+18), pk_wp[l], 256, 1024);
        repack_kn_k<<<(256*1024)/256, 256, 0, stream>>>(P(LB+20), pk_wg[l], 256, 1024);
        repack_kn_k<<<(1024*256)/256, 256, 0, stream>>>(P(LB+22), pk_wout[l], 1024, 256);
    }
    repack_roconv_k<<<1024, 256, 0, stream>>>(P(RO+0), pk_roconv);
    repack_kn_k<<<(256*256)/256, 256, 0, stream>>>(P(RO+2), pk_w1, 256, 256);

    // ---- recurrent rounds ----
    float* evtCur = evtA;
    float* evtPrev = evtB;
    const dim3 blk128(128);
    for (int r = 0; r < Rr; ++r) {
        embed_k<<<MROWS, 256, 0, stream>>>(inputs, fmask, P(3), P(4), P(5),
                                           P(6), P(7), P(8), P(9),
                                           state, evtCur, r);
        for (int l = 0; l < 3; ++l) {
            int LB = 10 + 30*l;
            // s = LN1(state)
            ln_k<<<MROWS, 256, 0, stream>>>(state, nullptr, P(LB+24), P(LB+25), bufS);
            // dilated conv stack as one implicit GEMM (K = 27*256)
            gemm_k<1,0><<<dim3(MROWS/16, 1), blk128, 0, stream>>>(
                bufS, pk_conv[l], cbias + l*HIDc, nullptr, bufT, MROWS, 256, 6912);
            // s = LN3(s + conv)
            ln_k<<<MROWS, 256, 0, stream>>>(bufS, bufT, P(LB+28), P(LB+29), bufS2);
            // qkv
            gemm_k<0,0><<<dim3(MROWS/16, 3), blk128, 0, stream>>>(
                bufS2, pk_qkv[l], P(LB+13), nullptr, qkvb, MROWS, 768, 256);
            // attention (scores + bias + softmax + attn@v), per (b,h)
            attn_k<<<Bsz*HEADSc, blk128, 0, stream>>>(
                qkvb, evtCur, evtPrev, sbh + (size_t)l*Ss*Ss*HEADSc,
                P(LB+16), P(LB+17), bufT);
            // out @ Wo
            gemm_k<0,0><<<dim3(MROWS/16, 1), blk128, 0, stream>>>(
                bufT, pk_wo[l], P(LB+15), nullptr, bufU, MROWS, 256, 256);
            // s = LN2(s + attn_proj)
            ln_k<<<MROWS, 256, 0, stream>>>(bufS2, bufU, P(LB+26), P(LB+27), bufS);
            // GLU FFN: g, then p * sigmoid(g), then Wout
            gemm_k<0,0><<<dim3(MROWS/16, 4), blk128, 0, stream>>>(
                bufS, pk_wg[l], P(LB+21), nullptr, gbuf, MROWS, 1024, 256);
            gemm_k<0,2><<<dim3(MROWS/16, 4), blk128, 0, stream>>>(
                bufS, pk_wp[l], P(LB+19), gbuf, hbuf, MROWS, 1024, 256);
            gemm_k<0,0><<<dim3(MROWS/16, 1), blk128, 0, stream>>>(
                hbuf, pk_wout[l], P(LB+23), nullptr, bufU, MROWS, 256, 1024);
            // state = LN3(s + ff)
            ln_k<<<MROWS, 256, 0, stream>>>(bufS, bufU, P(LB+28), P(LB+29), state);
        }
        // swap evt ping-pong
        float* tmp = evtPrev; evtPrev = evtCur; evtCur = tmp;
    }

    // ---- readout ----
    gemm_k<2,0><<<dim3(MROWS/16, 1), blk128, 0, stream>>>(
        state, pk_roconv, P(RO+1), nullptr, bufT, MROWS, 256, 1024);
    lines_k<<<Bsz*Gg, 256, 0, stream>>>(bufT, basis, linesb);
    gemm_k<0,1><<<dim3((Bsz*Gg)/16, 1), blk128, 0, stream>>>(
        linesb, pk_w1, P(RO+3), nullptr, y1b, Bsz*Gg, 256, 256);
    final_k<<<Bsz, 256, 0, stream>>>(y1b, P(RO+4), P(RO+5), (float*)d_out);
}
